// AttentionLayer_41970420418162
// MI455X (gfx1250) — matrix operand order repaired and checked by reading, not compiled, not run on hardware
//
#include <hip/hip_runtime.h>
#include <hip/hip_bf16.h>

typedef __attribute__((ext_vector_type(16))) __bf16 v16bf;
typedef __attribute__((ext_vector_type(8)))  float  v8f;

// Padded LDS layout (all row strides 16B-aligned, bank-conflict-free):
//  Q  : 32 rows  x 1040 B   (512 bf16 + 8 pad)
//  K  : 64 rows  x 1040 B
//  VT : 512 rows x 144 B    (64 bf16 + 8 pad)
//  S  : 2 grp x 16 rows x 272 B (64 f32 + 4 pad dwords)
//  P  : 8 waves x 16 rows x 144 B
//  M  : 32 rows x 272 B     (64 i32 + pad)
#define LDS_Q     0
#define LDS_K     33280
#define LDS_VT    99840
#define LDS_S     173568
#define LDS_P     182272
#define LDS_M     200704
#define LDS_TOTAL 209408

__device__ __forceinline__ unsigned short f2bf(float f) {
  unsigned u = __float_as_uint(f);
  u += 0x7FFFu + ((u >> 16) & 1u);          // round-to-nearest-even
  return (unsigned short)(u >> 16);
}

#if defined(__has_builtin) && __has_builtin(__builtin_amdgcn_cvt_pk_bf16_f32)
typedef __attribute__((ext_vector_type(2))) __bf16 v2bf;
__device__ __forceinline__ unsigned pk2(float a, float b) {
  union { v2bf v; unsigned u; } c;
  c.v = __builtin_amdgcn_cvt_pk_bf16_f32(a, b);
  return c.u;
}
#else
__device__ __forceinline__ unsigned pk2(float a, float b) {
  return (unsigned)f2bf(a) | ((unsigned)f2bf(b) << 16);
}
#endif

union ABCast { uint4 q[2]; v16bf v; };
// A-operand (16-bit 16x32): lanes<16 K0..7|K16..23, lanes>=16 K8..15|K24..31
__device__ __forceinline__ v16bf ld_a(const char* p) {
  ABCast u; u.q[0] = *(const uint4*)p; u.q[1] = *(const uint4*)(p + 32); return u.v;
}
// B-operand (16-bit 32x16): lanes<16 K0..7|K16..23, lanes>=16 K8..15|K24..31
__device__ __forceinline__ v16bf ld_b(const char* p) {
  ABCast u; u.q[0] = *(const uint4*)p; u.q[1] = *(const uint4*)(p + 32); return u.v;
}

// Async global->LDS copy of 16 bytes per lane (tracked by ASYNCcnt).
__device__ __forceinline__ void async_copy_b128(unsigned lds_off, const void* gptr) {
  asm volatile("global_load_async_to_lds_b128 %0, %1, off"
               :: "v"(lds_off), "v"(gptr) : "memory");
}
__device__ __forceinline__ void wait_async0() {
  asm volatile("s_wait_asynccnt 0x0" ::: "memory");
}

__global__ __launch_bounds__(256)
void fa_fwd_kernel(const float* __restrict__ Qg, const float* __restrict__ Kg,
                   const float* __restrict__ Vg, const int* __restrict__ Mg,
                   float* __restrict__ Og)
{
  extern __shared__ char smem[];
  const int tid  = threadIdx.x;
  const int wave = tid >> 5;
  const int lane = tid & 31;
  const int hlf  = lane >> 4;
  const int l15  = lane & 15;
  const int r    = wave >> 2;       // row group (16 q-rows)
  const int t    = wave & 3;        // S key-tile / PV d-chunk
  const int b    = blockIdx.x >> 6;
  const int qb   = (blockIdx.x & 63) * 32;

  // ---- Q tile (32x512) fp32 -> bf16 LDS (padded rows) ----
  {
    const float* Qbase = Qg + ((size_t)(b * 2048 + qb)) * 512;
    #pragma unroll 4
    for (int it = 0; it < 16; ++it) {
      int idx = (it * 256 + tid) * 4;
      int row = idx >> 9, col = idx & 511;
      float4 v = *(const float4*)(Qbase + row * 512 + col);
      uint2 p; p.x = pk2(v.x, v.y); p.y = pk2(v.z, v.w);
      *(uint2*)(smem + LDS_Q + row * 1040 + col * 2) = p;
    }
  }

  v8f O[8];
  float m[8], l[8];
  #pragma unroll
  for (int n = 0; n < 8; ++n) O[n] = {};
  #pragma unroll
  for (int i = 0; i < 8; ++i) { m[i] = -3.0e38f; l[i] = 0.f; }

  for (int kb = 0; kb < 2048; kb += 64) {
    __syncthreads();  // previous iteration fully consumed LDS

    // ---- async mask block (32 q-rows x 64 keys i32) -> LDS ----
    {
      #pragma unroll
      for (int j = 0; j < 2; ++j) {
        int id  = j * 256 + tid;        // 512 x 16B transfers
        int row = id >> 4, c16 = id & 15;
        const int* gp = Mg + ((long)(b * 2048 + qb + row)) * 2048 + kb + c16 * 4;
        unsigned loff = (unsigned)(unsigned long long)(smem + LDS_M + row * 272 + c16 * 16);
        async_copy_b128(loff, gp);
      }
    }
    // ---- K block (64x512) -> LDS bf16 ----
    const float* Kbase = Kg + ((size_t)(b * 2048 + kb)) * 512;
    #pragma unroll 4
    for (int it = 0; it < 32; ++it) {
      int idx = (it * 256 + tid) * 4;
      int row = idx >> 9, col = idx & 511;
      float4 v = *(const float4*)(Kbase + row * 512 + col);
      uint2 p; p.x = pk2(v.x, v.y); p.y = pk2(v.z, v.w);
      *(uint2*)(smem + LDS_K + row * 1040 + col * 2) = p;
    }
    // ---- V block (64x512) -> LDS bf16 TRANSPOSED ----
    const float* Vbase = Vg + ((size_t)(b * 2048 + kb)) * 512;
    #pragma unroll 2
    for (int it = 0; it < 8; ++it) {
      int tile = it * 256 + tid;
      int r0 = (tile >> 7) * 4, c0 = (tile & 127) * 4;
      float4 a0 = *(const float4*)(Vbase + (r0 + 0) * 512 + c0);
      float4 a1 = *(const float4*)(Vbase + (r0 + 1) * 512 + c0);
      float4 a2 = *(const float4*)(Vbase + (r0 + 2) * 512 + c0);
      float4 a3 = *(const float4*)(Vbase + (r0 + 3) * 512 + c0);
      uint2 p;
      p.x = pk2(a0.x, a1.x); p.y = pk2(a2.x, a3.x);
      *(uint2*)(smem + LDS_VT + (c0 + 0) * 144 + r0 * 2) = p;
      p.x = pk2(a0.y, a1.y); p.y = pk2(a2.y, a3.y);
      *(uint2*)(smem + LDS_VT + (c0 + 1) * 144 + r0 * 2) = p;
      p.x = pk2(a0.z, a1.z); p.y = pk2(a2.z, a3.z);
      *(uint2*)(smem + LDS_VT + (c0 + 2) * 144 + r0 * 2) = p;
      p.x = pk2(a0.w, a1.w); p.y = pk2(a2.w, a3.w);
      *(uint2*)(smem + LDS_VT + (c0 + 3) * 144 + r0 * 2) = p;
    }
    // ---- prefetch next K/V block into cache hierarchy ----
    if (kb + 64 < 2048) {
      const char* nk = (const char*)(Kg + ((size_t)(b * 2048 + kb + 64)) * 512);
      const char* nv = (const char*)(Vg + ((size_t)(b * 2048 + kb + 64)) * 512);
      #pragma unroll
      for (int j = 0; j < 4; ++j) {
        __builtin_prefetch(nk + (tid * 4 + j) * 128, 0, 3);
        __builtin_prefetch(nv + (tid * 4 + j) * 128, 0, 3);
      }
    }
    wait_async0();
    __syncthreads();

    // ---- S tile: rows 16r.. x keys kb+16t.., full D=512 reduction ----
    v8f s = {};
    #pragma unroll
    for (int ss = 0; ss < 16; ++ss) {
      v16bf A  = ld_a(smem + LDS_Q + (16 * r + l15) * 1040 + 64 * ss + 16 * hlf);
      v16bf Bm = ld_b(smem + LDS_K + (16 * t + l15) * 1040 + 64 * ss + 16 * hlf);
      s = __builtin_amdgcn_wmma_f32_16x16x32_bf16(false, A, false, Bm, (short)0, s,
                                                  false, false);
    }
    #pragma unroll
    for (int i = 0; i < 8; ++i)
      *(float*)(smem + LDS_S + r * 4352 + (i + 8 * hlf) * 272 + (16 * t + l15) * 4) = s[i];
    __syncthreads();

    // ---- read full 16x64 S + mask, leaky-relu + online softmax ----
    float e[32];
    #pragma unroll
    for (int tt = 0; tt < 4; ++tt)
      #pragma unroll
      for (int i = 0; i < 8; ++i) {
        float x = *(const float*)(smem + LDS_S + r * 4352 + (i + 8 * hlf) * 272 +
                                  (16 * tt + l15) * 4);
        int mv = *(const int*)(smem + LDS_M + (16 * r + i + 8 * hlf) * 272 +
                               (16 * tt + l15) * 4);
        x = (x >= 0.f) ? x : 0.2f * x;            // leaky relu
        e[tt * 8 + i] = mv ? x : -1.0e9f;         // mask
      }
    float al[8], rs[8];
    #pragma unroll
    for (int i = 0; i < 8; ++i) {
      float mx = fmaxf(fmaxf(e[i], e[8 + i]), fmaxf(e[16 + i], e[24 + i]));
      #pragma unroll
      for (int d = 1; d < 16; d <<= 1) mx = fmaxf(mx, __shfl_xor(mx, d, 32));
      float mn = fmaxf(m[i], mx);
      al[i] = __expf(m[i] - mn);
      m[i] = mn;
      rs[i] = 0.f;
    }
    #pragma unroll
    for (int tt = 0; tt < 4; ++tt)
      #pragma unroll
      for (int i = 0; i < 8; ++i) {
        float p = __expf(e[tt * 8 + i] - m[i]);
        e[tt * 8 + i] = p;
        rs[i] += p;
      }
    #pragma unroll
    for (int i = 0; i < 8; ++i) {
      #pragma unroll
      for (int d = 1; d < 16; d <<= 1) rs[i] += __shfl_xor(rs[i], d, 32);
      l[i] = l[i] * al[i] + rs[i];
    }
    #pragma unroll
    for (int n = 0; n < 8; ++n)
      #pragma unroll
      for (int i = 0; i < 8; ++i) O[n][i] *= al[i];

    // ---- P (C layout) -> per-wave scratch -> reload in A layout ----
    char* ps = smem + LDS_P + wave * 2304;
    #pragma unroll
    for (int tt = 0; tt < 4; ++tt)
      #pragma unroll
      for (int i = 0; i < 8; ++i)
        *(unsigned short*)(ps + (i + 8 * hlf) * 144 + (16 * tt + l15) * 2) =
            f2bf(e[tt * 8 + i]);
    v16bf P0 = ld_a(ps + l15 * 144 + 0  + 16 * hlf);
    v16bf P1 = ld_a(ps + l15 * 144 + 64 + 16 * hlf);

    // ---- O += P x V (this wave's 128-column d-chunk) ----
    #pragma unroll
    for (int n = 0; n < 8; ++n) {
      const char* vb = smem + LDS_VT + (128 * t + 16 * n + l15) * 144;
      v16bf B0 = ld_b(vb + 0  + 16 * hlf);
      v16bf B1 = ld_b(vb + 64 + 16 * hlf);
      O[n] = __builtin_amdgcn_wmma_f32_16x16x32_bf16(false, P0, false, B0, (short)0,
                                                     O[n], false, false);
      O[n] = __builtin_amdgcn_wmma_f32_16x16x32_bf16(false, P1, false, B1, (short)0,
                                                     O[n], false, false);
    }
  }

  // ---- epilogue: normalize and store ----
  #pragma unroll
  for (int n = 0; n < 8; ++n)
    #pragma unroll
    for (int i = 0; i < 8; ++i) {
      long row = (long)b * 2048 + qb + 16 * r + i + 8 * hlf;
      Og[row * 512 + 128 * t + 16 * n + l15] = O[n][i] / l[i];
    }
}

extern "C" void kernel_launch(void* const* d_in, const int* in_sizes, int n_in,
                              void* d_out, int out_size, void* d_ws, size_t ws_size,
                              hipStream_t stream) {
  const float* Q = (const float*)d_in[0];
  const float* K = (const float*)d_in[1];
  const float* V = (const float*)d_in[2];
  const int*   M = (const int*)d_in[3];
  float* O = (float*)d_out;

  hipFuncSetAttribute((const void*)fa_fwd_kernel,
                      hipFuncAttributeMaxDynamicSharedMemorySize, LDS_TOTAL);
  // 8 batches x 64 q-tiles of 32 rows = 512 workgroups, 8 waves each
  fa_fwd_kernel<<<dim3(512), dim3(256), LDS_TOTAL, stream>>>(Q, K, V, M, O);
}
